// Gate_18004502905040
// MI455X (gfx1250) — matrix operand order repaired and checked by reading, not compiled, not run on hardware
//
#include <hip/hip_runtime.h>
#include <stdint.h>

// ---------------------------------------------------------------------------
// MoE gate (DeepSeek router) for MI455X / gfx1250.
//   A) weight split fp32 -> bf16 hi/lo planes (once, tiny)
//   B) score GEMM: bf16 WMMA 16x16x32, hi/lo split-bf16 compensation
//      (3 wmma per tile -> near-fp32 accuracy), double-buffered static LDS,
//      async global->LDS weight staging, sigmoid epilogue to ws
//   C) grouped top-2 / top-4-group / top-8-expert routing
// Roofline: x is 470 MB read once -> 20 us floor @ 23.3 TB/s; bf16 WMMA path
// keeps compute (~180 GFLOP-equiv) at or below that floor.
// ---------------------------------------------------------------------------

typedef __attribute__((ext_vector_type(16))) __bf16 v16bf;
typedef __attribute__((ext_vector_type(8)))  __bf16 v8bf;
typedef __attribute__((ext_vector_type(4)))  __bf16 v4bf;
typedef __attribute__((ext_vector_type(8)))  float  v8f;
typedef __attribute__((ext_vector_type(4)))  int    v4i;

#define D_DIM   7168
#define E_DIM   256
#define KCHUNK  32
#define NCHUNK  (D_DIM / KCHUNK)   // 224
#define BM      64                 // tokens per GEMM block
#define BN      128                // experts per GEMM block
#define WS      40                 // LDS row stride (bf16), conflict-free + 16B aligned
#define XS      40
#define RS      260                // routing LDS row stride (f32), 16B aligned rows
#define N_GROUPS     8
#define GROUP_SIZE   32
#define TOPK_GROUPS  4
#define TOPK         8
#define ROUTE_SCALE  2.5f

#if defined(__has_builtin)
#if __has_builtin(__builtin_amdgcn_global_load_async_to_lds_b128) && \
    __has_builtin(__builtin_amdgcn_s_wait_asynccnt)
#define USE_ASYNC 1
#endif
#endif
#ifndef USE_ASYNC
#define USE_ASYNC 0
#endif

// addrspace-typed int4 pointees for the async builtin (HIP prints AS1 as
// "__device__", AS3 as "__shared__" in diagnostics)
typedef __attribute__((address_space(1))) v4i as1_v4i;
typedef __attribute__((address_space(3))) v4i as3_v4i;

#if USE_ASYNC
static __device__ __forceinline__ void async_cp16(void* lds_dst, const void* gsrc) {
    __builtin_amdgcn_global_load_async_to_lds_b128(
        (as1_v4i*)(uintptr_t)gsrc, (as3_v4i*)lds_dst, 0, 0);
}
#endif

static __device__ __forceinline__ void split_bf16(float f, __bf16& h, __bf16& l) {
    __bf16 hh = (__bf16)f;           // RNE to bf16
    h = hh;
    l = (__bf16)(f - (float)hh);     // residual
}

// A fragment (16-bit A 16x32): lanes 0-15: K {0..7,16..23}; lanes 16-31: K {8..15,24..31}
static __device__ __forceinline__ v16bf ld_frag_a(const __bf16* rowp, int kbase) {
    v8bf a = *(const v8bf*)(rowp + kbase);
    v8bf b = *(const v8bf*)(rowp + kbase + 16);
    v16bf r;
#pragma unroll
    for (int i = 0; i < 8; ++i) { r[i] = a[i]; r[8 + i] = b[i]; }
    return r;
}

// B fragment (16-bit B 32x16): lane holds one expert column; lanes 0-15: K {0..7,16..23}; lanes 16-31: K {8..15,24..31}
static __device__ __forceinline__ v16bf ld_frag_b(const __bf16* p) {
    v8bf a = *(const v8bf*)p;
    v8bf b = *(const v8bf*)(p + 16);
    v16bf r;
#pragma unroll
    for (int i = 0; i < 8; ++i) { r[i] = a[i]; r[8 + i] = b[i]; }
    return r;
}

// ---- kernel A: weight split ------------------------------------------------

__global__ void Gate_wsplit_kernel(const float* __restrict__ w,
                                   __bf16* __restrict__ wh,
                                   __bf16* __restrict__ wl, int n) {
    int i = blockIdx.x * blockDim.x + threadIdx.x;
    if (i < n) {
        float f = w[i];
        __bf16 h, l;
        split_bf16(f, h, l);
        wh[i] = h;
        wl[i] = l;
    }
}

// ---- kernel B: score GEMM + sigmoid ---------------------------------------

__global__ __launch_bounds__(512) void Gate_gemm_kernel(
    const float* __restrict__ x, const __bf16* __restrict__ whg,
    const __bf16* __restrict__ wlg, float* __restrict__ scores) {
    __shared__ __bf16 whL[2][BN * WS];   // 2 x 10240 B
    __shared__ __bf16 wlL[2][BN * WS];   // 2 x 10240 B
    __shared__ __bf16 xhL[2][BM * XS];   // 2 x 5120 B
    __shared__ __bf16 xlL[2][BM * XS];   // 2 x 5120 B  -> 60 KB total

    const int tid   = threadIdx.x;
    const int tBase = blockIdx.x * BM;
    const int nBase = blockIdx.y * BN;
    const int wave  = tid >> 5;
    const int lane  = tid & 31;
    const int mt    = wave >> 2;           // 0..3 : 16-token tile
    const int np    = wave & 3;            // 0..3 : pair of expert tiles
    const int row   = lane & 15;
    const int kbaseA = (lane < 16) ? 0 : 8;
    const int koffB  = (lane < 16) ? 0 : 8;
    const int e0     = np * 32 + row;      // local expert, N-tile 2*np
    const int e1     = e0 + 16;            // local expert, N-tile 2*np+1

    v8f acc0 = {};
    v8f acc1 = {};

    auto stage = [&](int b, int k0) {
        // weight tile: 128 experts x 32 K bf16, hi + lo planes (pure copies)
        const int we    = tid >> 2;        // 0..127
        const int wpart = tid & 3;         // 0..3 : 16B chunk in 64B row segment
#pragma unroll
        for (int pl = 0; pl < 2; ++pl) {
            const __bf16* gp = pl ? wlg : whg;
            __bf16* lp = pl ? &wlL[b][0] : &whL[b][0];
            const __bf16* src = gp + (size_t)(nBase + we) * D_DIM + k0 + wpart * 8;
            __bf16* dst = lp + we * WS + wpart * 8;
#if USE_ASYNC
            async_cp16(dst, src);
#else
            *(uint4*)dst = *(const uint4*)src;
#endif
        }
        // x tile: 64 tokens x 32 K fp32 -> hi/lo bf16
        {
            const int t    = tid >> 3;     // 0..63
            const int part = tid & 7;      // 0..7
            float4 v = *(const float4*)(x + (size_t)(tBase + t) * D_DIM + k0 + part * 4);
            __bf16 h0, h1, h2, h3, l0, l1, l2, l3;
            split_bf16(v.x, h0, l0);
            split_bf16(v.y, h1, l1);
            split_bf16(v.z, h2, l2);
            split_bf16(v.w, h3, l3);
            v4bf hv = {h0, h1, h2, h3};
            v4bf lv = {l0, l1, l2, l3};
            *(v4bf*)&xhL[b][t * XS + part * 4] = hv;
            *(v4bf*)&xlL[b][t * XS + part * 4] = lv;
        }
    };

    stage(0, 0);
#if USE_ASYNC
    __builtin_amdgcn_s_wait_asynccnt(0);
#endif
    __syncthreads();

    for (int kc = 0; kc < NCHUNK; ++kc) {
        const int c = kc & 1;
        if (kc + 1 < NCHUNK) stage(c ^ 1, (kc + 1) * KCHUNK);

        const __bf16* xrh = &xhL[c][(mt * 16 + row) * XS];
        const __bf16* xrl = &xlL[c][(mt * 16 + row) * XS];
        v16bf ah = ld_frag_a(xrh, kbaseA);
        v16bf al = ld_frag_a(xrl, kbaseA);

        v16bf bh0 = ld_frag_b(&whL[c][e0 * WS + koffB]);
        v16bf bl0 = ld_frag_b(&wlL[c][e0 * WS + koffB]);
        acc0 = __builtin_amdgcn_wmma_f32_16x16x32_bf16(false, ah, false, bh0,
                                                       (short)0, acc0, false, false);
        acc0 = __builtin_amdgcn_wmma_f32_16x16x32_bf16(false, ah, false, bl0,
                                                       (short)0, acc0, false, false);
        acc0 = __builtin_amdgcn_wmma_f32_16x16x32_bf16(false, al, false, bh0,
                                                       (short)0, acc0, false, false);

        v16bf bh1 = ld_frag_b(&whL[c][e1 * WS + koffB]);
        v16bf bl1 = ld_frag_b(&wlL[c][e1 * WS + koffB]);
        acc1 = __builtin_amdgcn_wmma_f32_16x16x32_bf16(false, ah, false, bh1,
                                                       (short)0, acc1, false, false);
        acc1 = __builtin_amdgcn_wmma_f32_16x16x32_bf16(false, ah, false, bl1,
                                                       (short)0, acc1, false, false);
        acc1 = __builtin_amdgcn_wmma_f32_16x16x32_bf16(false, al, false, bh1,
                                                       (short)0, acc1, false, false);

#if USE_ASYNC
        __builtin_amdgcn_s_wait_asynccnt(0);
#endif
        __syncthreads();
    }

    // sigmoid epilogue: C/D layout: VGPR j, lane L -> M = j + (L<16?0:8), N = L&15
    const int rofs = (lane >= 16) ? 8 : 0;
    const int col  = nBase + np * 32 + (lane & 15);
#pragma unroll
    for (int j = 0; j < 8; ++j) {
        int tok = tBase + mt * 16 + j + rofs;
        scores[(size_t)tok * E_DIM + col]      = 1.0f / (1.0f + __expf(-acc0[j]));
        scores[(size_t)tok * E_DIM + col + 16] = 1.0f / (1.0f + __expf(-acc1[j]));
    }
}

// ---- kernel C: grouped top-k routing --------------------------------------

__global__ __launch_bounds__(128) void Gate_route_kernel(
    const float* __restrict__ scores, const float* __restrict__ bias_g,
    float* __restrict__ out_w, int* __restrict__ out_i) {
    __shared__ float srow[32 * RS];     // 33280 B
    __shared__ float bias_s[E_DIM];     // 1024 B

    const int tid   = threadIdx.x;
    const int tBase = blockIdx.x * 32;

    for (int i = tid; i < E_DIM; i += 128) bias_s[i] = bias_g[i];

    // stage 32 score rows, fully coalesced float4 loads
#pragma unroll
    for (int r = 0; r < 16; ++r) {
        int idx = tid + r * 128;        // 0..2047
        int row = idx >> 6;             // 0..31
        int ch  = idx & 63;             // 0..63 (x4 floats)
        float4 v = *(const float4*)(scores + (size_t)(tBase + row) * E_DIM + ch * 4);
        *(float4*)&srow[row * RS + ch * 4] = v;   // RS=260 -> rows stay 16B aligned
    }
    __syncthreads();

    if (tid < 32) {
        const float* rowp = &srow[tid * RS];

        // group score = sum of top-2 (sigmoid + bias) per group of 32
        float gs[N_GROUPS];
#pragma unroll
        for (int g = 0; g < N_GROUPS; ++g) {
            float m1 = -3.0e38f, m2 = -3.0e38f;
            for (int i = 0; i < GROUP_SIZE; ++i) {
                int e   = g * GROUP_SIZE + i;
                float v = rowp[e] + bias_s[e];
                if (v > m1)      { m2 = m1; m1 = v; }
                else if (v > m2) { m2 = v; }
            }
            gs[g] = m1 + m2;
        }
        // top-4 groups (stable: lower index wins ties)
        unsigned keep = 0;
#pragma unroll
        for (int r = 0; r < TOPK_GROUPS; ++r) {
            float best = -3.0e38f; int bi = 0;
            for (int g = 0; g < N_GROUPS; ++g)
                if (!((keep >> g) & 1u) && gs[g] > best) { best = gs[g]; bi = g; }
            keep |= 1u << bi;
        }
        // top-8 experts over kept groups (stable descending insertion)
        float tv[TOPK]; int ti[TOPK];
#pragma unroll
        for (int i = 0; i < TOPK; ++i) { tv[i] = -3.0e38f; ti[i] = 0; }
        for (int g = 0; g < N_GROUPS; ++g) {
            if (!((keep >> g) & 1u)) continue;
            for (int i = 0; i < GROUP_SIZE; ++i) {
                int e   = g * GROUP_SIZE + i;
                float v = rowp[e] + bias_s[e];
                if (v > tv[TOPK - 1]) {
                    int pos = TOPK - 1;
                    while (pos > 0 && v > tv[pos - 1]) {
                        tv[pos] = tv[pos - 1]; ti[pos] = ti[pos - 1]; --pos;
                    }
                    tv[pos] = v; ti[pos] = e;
                }
            }
        }
        // weights from ORIGINAL sigmoid scores, normalized, scaled
        float w[TOPK]; float s = 0.0f;
#pragma unroll
        for (int i = 0; i < TOPK; ++i) { w[i] = rowp[ti[i]]; s += w[i]; }
        float scale = ROUTE_SCALE / s;
        int tok = tBase + tid;
#pragma unroll
        for (int i = 0; i < TOPK; ++i) {
            out_w[(size_t)tok * TOPK + i] = w[i] * scale;
            out_i[(size_t)tok * TOPK + i] = ti[i];
        }
    }
}

// ---------------------------------------------------------------------------

extern "C" void kernel_launch(void* const* d_in, const int* in_sizes, int n_in,
                              void* d_out, int out_size, void* d_ws,
                              size_t ws_size, hipStream_t stream) {
    const float* x      = (const float*)d_in[0];
    const float* weight = (const float*)d_in[1];
    const float* bias   = (const float*)d_in[2];

    const int T  = in_sizes[0] / D_DIM;     // 16384
    const int NW = E_DIM * D_DIM;           // 1,835,008

    __bf16* wh      = (__bf16*)d_ws;
    __bf16* wl      = wh + (size_t)NW;
    float*  scoresW = (float*)(wl + (size_t)NW);   // [T][256] f32

    hipLaunchKernelGGL(Gate_wsplit_kernel, dim3((NW + 255) / 256), dim3(256), 0,
                       stream, weight, wh, wl, NW);

    hipLaunchKernelGGL(Gate_gemm_kernel, dim3(T / BM, E_DIM / BN), dim3(512), 0,
                       stream, x, wh, wl, scoresW);

    float* out_w = (float*)d_out;
    int*   out_i = (int*)d_out + (size_t)T * TOPK;
    hipLaunchKernelGGL(Gate_route_kernel, dim3(T / 32), dim3(128), 0, stream,
                       scoresW, bias, out_w, out_i);
}
